// MixtureOfUnityExperts_16690242912674
// MI455X (gfx1250) — matrix operand order repaired and checked by reading, not compiled, not run on hardware
//
#include <hip/hip_runtime.h>
#include <hip/hip_bf16.h>
#include <math.h>

typedef __attribute__((ext_vector_type(16))) _Float16 v16h;
typedef __attribute__((ext_vector_type(8)))  float    v8f;

#define HDIM   256
#define MTILE  64
#define TPB    128          // 4 wave32 waves
#define BUFH   (128 * 256)  // halves per staging buffer (64KB)
#define SQRT_PHI 1.2720196495140689643f

// ---------------- WMMA fragment loaders (ISA 7.12.2 layouts) ----------------
// All addresses are 16B-aligned by construction -> ds_load_b128 pairs.
__device__ __forceinline__ v16h ld_a_frag(const _Float16* base, int row0, int rs,
                                          int k0, int lane) {
  int r = row0 + (lane & 15);
  int koff = k0 + ((lane & 16) ? 8 : 0);
  const _Float16* p = base + r * rs + koff;
  union { v16h v; uint4 q[2]; } f;
  f.q[0] = *(const uint4*)p;
  f.q[1] = *(const uint4*)(p + 16);
  return f.v;
}

__device__ __forceinline__ v16h ld_b_frag(const _Float16* base, int n0, int ks,
                                          int k0, int lane) {
  int n = n0 + (lane & 15);
  int koff = k0 + ((lane & 16) ? 8 : 0);
  const _Float16* p = base + n * ks + koff;
  union { v16h v; uint4 q[2]; } f;
  f.q[0] = *(const uint4*)p;
  f.q[1] = *(const uint4*)(p + 16);
  return f.v;
}

// Software-pipelined WMMA tile block: prefetches the next A/B fragment before
// each WMMA so ds_load latency overlaps matrix issue (critical at 1 wave/SIMD).
// ksteps (<=8) and ntiles (<=8) constant-fold at every call site.
__device__ __forceinline__ void wmma_block(const _Float16* ab, int am0, int ars,
                                           int ak0, const _Float16* bb, int bks,
                                           int ksteps, int ntiles, v8f* acc,
                                           int lane) {
  v16h af = ld_a_frag(ab, am0, ars, ak0, lane);
  v16h bf = ld_b_frag(bb, 0, bks, 0, lane);
#pragma unroll
  for (int kk = 0; kk < 8; ++kk) {
    if (kk < ksteps) {
      const int k0 = kk * 32;
      v16h afn = af;
      if (kk + 1 < ksteps) afn = ld_a_frag(ab, am0, ars, ak0 + k0 + 32, lane);
#pragma unroll
      for (int t = 0; t < 8; ++t) {
        if (t < ntiles) {
          v16h nb = bf;
          if (t + 1 < ntiles)       nb = ld_b_frag(bb, (t + 1) * 16, bks, k0, lane);
          else if (kk + 1 < ksteps) nb = ld_b_frag(bb, 0, bks, k0 + 32, lane);
          acc[t] = __builtin_amdgcn_wmma_f32_16x16x32_f16(
              false, af, false, bf, (short)0, acc[t], false, false);
          bf = nb;
        }
      }
      af = afn;
    }
  }
}

__device__ __forceinline__ float gelu_exact(float x) {
  return 0.5f * x * (1.0f + erff(x * 0.70710678118654752440f));
}
__device__ __forceinline__ float sigmoidf(float x) { return 1.0f / (1.0f + expf(-x)); }

// ---- CDNA5 async global->LDS copy (ASYNCcnt path), 16B per lane ----
__device__ __forceinline__ void cp_async16(_Float16* lds_dst, const _Float16* gsrc) {
  unsigned ldsoff = (unsigned)(unsigned long long)(void*)lds_dst;  // low 32b = LDS addr
  unsigned long long ga = (unsigned long long)gsrc;
  asm volatile("global_load_async_to_lds_b128 %0, %1, off"
               :: "v"(ldsoff), "v"(ga) : "memory");
}
__device__ __forceinline__ void async_wait_0() {
  asm volatile("s_wait_asynccnt 0" ::: "memory");
}
__device__ __forceinline__ void async_wait_32() {  // <= one 32-op chunk in flight
  asm volatile("s_wait_asynccnt 32" ::: "memory");
}

// Contiguous f16 block copy global->LDS via async DMA (nhalves % 1024 == 0).
__device__ __forceinline__ void stage_copy(_Float16* __restrict__ dst,
                                           const _Float16* __restrict__ src,
                                           int nhalves, int tid) {
  for (int i = tid * 8; i < nhalves; i += TPB * 8) cp_async16(dst + i, src + i);
}

// ---------------- prep: one-time f32 -> padded/staging-ordered f16 ----------------
// [d1pad][256] row-major, rows >= d1 zero-filled.
__global__ void prep_w1_kernel(const float* __restrict__ src, _Float16* __restrict__ dst,
                               int d1, int d1pad) {
  int total = d1pad * HDIM;
  for (int i = blockIdx.x * blockDim.x + threadIdx.x; i < total;
       i += gridDim.x * blockDim.x) {
    int n = i >> 8;
    float v = (n < d1) ? src[n * HDIM + (i & (HDIM - 1))] : 0.0f;
    dst[i] = (_Float16)v;
  }
}
// w2h layout: per-256-K-chunk blocks, each [256][kc] contiguous, k >= d1 zero-filled.
__global__ void prep_w2_kernel(const float* __restrict__ src, _Float16* __restrict__ dst,
                               int d1, int d1pad) {
  int kc0 = d1pad < 256 ? d1pad : 256;
  int total = HDIM * d1pad;
  for (int i = blockIdx.x * blockDim.x + threadIdx.x; i < total;
       i += gridDim.x * blockDim.x) {
    int rem = i, kc = kc0, gkb = 0;
    if (i >= HDIM * kc0) { rem = i - HDIM * kc0; kc = d1pad - 256; gkb = 256; }
    int n = rem / kc;
    int gk = gkb + (rem - n * kc);
    float v = (gk < d1) ? src[(size_t)n * d1 + gk] : 0.0f;
    dst[i] = (_Float16)v;
  }
}

// ---------------- kernels ----------------
__global__ void zero_kernel(float* __restrict__ p, size_t n) {
  size_t i = (size_t)blockIdx.x * blockDim.x + threadIdx.x;
  size_t st = (size_t)gridDim.x * blockDim.x;
  for (; i < n; i += st) p[i] = 0.0f;
}

// wave-per-token gating: logits -> softmax(sqrt(phi)*l) -> top2 (normalized)
__global__ __launch_bounds__(256) void gate_kernel(
    const float* __restrict__ x, const float* __restrict__ gw,
    const float* __restrict__ gb, float* __restrict__ topg,
    int* __restrict__ idxo, int T) {
  int gwv = (int)((blockIdx.x * blockDim.x + threadIdx.x) >> 5);
  int lane = threadIdx.x & 31;
  if (gwv >= T) return;
  const float* xr = x + (size_t)gwv * HDIM;
  float acc[6] = {0.f, 0.f, 0.f, 0.f, 0.f, 0.f};
  for (int k = lane; k < HDIM; k += 32) {
    float xv = xr[k];
#pragma unroll
    for (int e = 0; e < 6; ++e) acc[e] += xv * gw[e * HDIM + k];
  }
#pragma unroll
  for (int off = 16; off >= 1; off >>= 1)
#pragma unroll
    for (int e = 0; e < 6; ++e) acc[e] += __shfl_xor(acc[e], off, 32);
  if (lane == 0) {
    float lg[6], mx = -1e30f;
#pragma unroll
    for (int e = 0; e < 6; ++e) { lg[e] = (acc[e] + gb[e]) * SQRT_PHI; mx = fmaxf(mx, lg[e]); }
    float s = 0.f;
#pragma unroll
    for (int e = 0; e < 6; ++e) { lg[e] = expf(lg[e] - mx); s += lg[e]; }
    float inv = 1.0f / s;
#pragma unroll
    for (int e = 0; e < 6; ++e) lg[e] *= inv;
    int i0 = 0;
#pragma unroll
    for (int e = 1; e < 6; ++e) if (lg[e] > lg[i0]) i0 = e;
    int i1 = (i0 == 0) ? 1 : 0;
#pragma unroll
    for (int e = 0; e < 6; ++e) if (e != i0 && lg[e] > lg[i1]) i1 = e;
    float g0 = lg[i0], g1 = lg[i1], ns = 1.0f / (g0 + g1);
    topg[2 * gwv + 0] = g0 * ns;
    topg[2 * gwv + 1] = g1 * ns;
    idxo[2 * gwv + 0] = i0;
    idxo[2 * gwv + 1] = i1;
  }
}

// Fused dense expert. Flattened chunk schedule, double-buffered async staging:
//   chunks [0,NCH1)           : GEMM1 weight chunks (w1h)
//   chunks [NCH1,NCH1+2*KCH)  : GEMM2 weight chunks (w2h)
//   chunk  NTOT-1             : confidence-head weights (cw1h)
template <int D1PAD, int KIND>
__global__ __launch_bounds__(TPB) void expert_kernel(
    const float* __restrict__ x, const float* __restrict__ topg,
    const int* __restrict__ idxp, const float* __restrict__ spec,
    const _Float16* __restrict__ w1h, const float* __restrict__ b1,
    const _Float16* __restrict__ w2h, const float* __restrict__ b2,
    const float* __restrict__ lng, const float* __restrict__ lnb,
    const _Float16* __restrict__ cw1h, const float* __restrict__ cb1,
    const float* __restrict__ cw2, const float* __restrict__ cb2,
    float* __restrict__ comb, float* __restrict__ confa, int d1, int eidx) {
  constexpr int NCH1 = (D1PAD + 127) / 128;
  constexpr int KCH  = (D1PAD + 255) / 256;
  constexpr int KC0  = D1PAD < 256 ? D1PAD : 256;
  constexpr int NTOT = NCH1 + 2 * KCH + 1;

  extern __shared__ char smem[];
  _Float16* sh_h = (_Float16*)smem;              // [64][256] f16
  _Float16* sh_a = sh_h + MTILE * HDIM;          // [64][D1PAD] f16 (later LN'd y f16)
  _Float16* sh_b = sh_a + MTILE * D1PAD;         // 2 x [128][256] f16 staging
  float*    sh_y = (float*)(sh_b + 2 * BUFH);    // [64][256] f32
  float*    sh_g = sh_y + MTILE * HDIM;          // [64]
  const int tid = threadIdx.x;
  const int lane = tid & 31;
  const int m0 = (tid >> 5) * 16;
  const long tok0 = (long)blockIdx.x * MTILE;
  const v8f vzero = {0.f, 0.f, 0.f, 0.f, 0.f, 0.f, 0.f, 0.f};

  auto issue = [&](int i) {
    const _Float16* src;
    if (i < NCH1) {
      src = w1h + i * 128 * HDIM;
    } else if (i < NCH1 + 2 * KCH) {
      int j = i - NCH1;
      int nbase = (j / KCH) * 128;
      int c = j % KCH;
      int kc = (D1PAD - c * 256) < 256 ? (D1PAD - c * 256) : 256;
      src = w2h + c * HDIM * KC0 + nbase * kc;
    } else {
      src = cw1h;
    }
    stage_copy(sh_b + (i & 1) * BUFH, src, BUFH, tid);  // uniform 32-op chunks
  };

  issue(0);
  issue(1);

  // load x tile + spec shift -> f16 (overlaps in-flight DMA)
  for (int i = tid * 4; i < MTILE * HDIM; i += TPB * 4) {
    int col = i & (HDIM - 1);
    float4 v = *(const float4*)(x + tok0 * HDIM + i);
    sh_h[i + 0] = (_Float16)(v.x + spec[col + 0]);
    sh_h[i + 1] = (_Float16)(v.y + spec[col + 1]);
    sh_h[i + 2] = (_Float16)(v.z + spec[col + 2]);
    sh_h[i + 3] = (_Float16)(v.w + spec[col + 3]);
  }

  v8f acc2[8];  // GEMM2 accumulators persist across its K chunks
#pragma unroll
  for (int t = 0; t < 8; ++t) acc2[t] = vzero;

#pragma unroll
  for (int i = 0; i < NTOT; ++i) {
    if (i + 1 < NTOT) async_wait_32(); else async_wait_0();
    __syncthreads();
    const _Float16* buf = sh_b + (i & 1) * BUFH;

    if (i < NCH1) {
      // ---- GEMM1 chunk: a = act1(h @ w1^T + b1) ----
      const int nbase = i * 128;
      const int nc = (D1PAD - nbase) < 128 ? (D1PAD - nbase) : 128;
      const int ntiles = nc >> 4;
      v8f acc[8];
#pragma unroll
      for (int t = 0; t < 8; ++t) acc[t] = vzero;
      wmma_block(sh_h, m0, HDIM, 0, buf, 256, 8, ntiles, acc, lane);
#pragma unroll
      for (int t = 0; t < 8; ++t) {
        if (t < ntiles) {
#pragma unroll
          for (int ii = 0; ii < 8; ++ii) {
            int m = m0 + ii + ((lane & 16) ? 8 : 0);
            int n = nbase + t * 16 + (lane & 15);
            float v = acc[t][ii] + (n < d1 ? b1[n] : 0.0f);
            if (KIND == 0)      v = fmaxf(v, 0.0f);
            else if (KIND == 3) v = tanhf(v);
            else                v = gelu_exact(v);
            sh_a[m * D1PAD + n] = (_Float16)v;
          }
        }
      }
    } else if (i < NCH1 + 2 * KCH) {
      // ---- GEMM2 chunk: y += a @ w2^T ----
      const int j = i - NCH1;
      const int nbase = (j / KCH) * 128;
      const int c = j % KCH;
      const int kbase = c * 256;
      const int kc = (D1PAD - kbase) < 256 ? (D1PAD - kbase) : 256;
      if (c == 0) {
#pragma unroll
        for (int t = 0; t < 8; ++t) acc2[t] = vzero;
      }
      wmma_block(sh_a, m0, D1PAD, kbase, buf, kc, kc >> 5, 8, acc2, lane);
      if (c == KCH - 1) {
#pragma unroll
        for (int t = 0; t < 8; ++t)
#pragma unroll
          for (int ii = 0; ii < 8; ++ii) {
            int m = m0 + ii + ((lane & 16) ? 8 : 0);
            int n = nbase + t * 16 + (lane & 15);
            float v = acc2[t][ii] + b2[n];
            if (KIND == 3)      v = tanhf(v);
            else if (KIND == 2) v = gelu_exact(v);
            sh_y[m * HDIM + n] = v;
          }
      }
    } else {
      // ---- LayerNorm + gate lookup, then WMMA confidence head ----
      if (tid < MTILE) {
        float* yr = sh_y + tid * HDIM;
        float s = 0.f;
        for (int jj = 0; jj < HDIM; ++jj) s += yr[jj];
        float mean = s * (1.0f / HDIM);
        float vs = 0.f;
        for (int jj = 0; jj < HDIM; ++jj) { float d = yr[jj] - mean; vs += d * d; }
        float inv = rsqrtf(vs * (1.0f / HDIM) + 1e-5f);
        for (int jj = 0; jj < HDIM; ++jj) {
          float yn = (yr[jj] - mean) * inv * lng[jj] + lnb[jj];
          yr[jj] = yn;
          sh_a[tid * HDIM + jj] = (_Float16)yn;  // A-matrix for conf GEMM
        }
        long tok = tok0 + tid;
        float g = 0.f;
        if (idxp[2 * tok] == eidx)          g = topg[2 * tok];
        else if (idxp[2 * tok + 1] == eidx) g = topg[2 * tok + 1];
        sh_g[tid] = g;
      }
      __syncthreads();
      v8f ca[4];
#pragma unroll
      for (int t = 0; t < 4; ++t) ca[t] = vzero;
      wmma_block(sh_a, m0, HDIM, 0, buf, 256, 8, 4, ca, lane);
#pragma unroll
      for (int ii = 0; ii < 8; ++ii) {
        float s = 0.f;
#pragma unroll
        for (int t = 0; t < 4; ++t) {
          int n = t * 16 + (lane & 15);
          s += fmaxf(ca[t][ii] + cb1[n], 0.0f) * cw2[n];
        }
        s += __shfl_xor(s, 1, 32);
        s += __shfl_xor(s, 2, 32);
        s += __shfl_xor(s, 4, 32);
        s += __shfl_xor(s, 8, 32);
        if ((lane & 15) == 0) {
          int m = m0 + ii + ((lane & 16) ? 8 : 0);
          confa[tok0 + m] += sh_g[m] * sigmoidf(s + cb2[0]);
        }
      }
    }
    __syncthreads();
    if (i + 2 < NTOT) issue(i + 2);
  }

  // ---- combined += g * y (only when this expert was selected) ----
  for (int i = tid * 4; i < MTILE * HDIM; i += TPB * 4) {
    int r = i >> 8;
    float g = sh_g[r];
    if (g != 0.f) {
      float4 yv = *(const float4*)(sh_y + i);
      float4* dst = (float4*)(comb + tok0 * HDIM + i);
      float4 o = *dst;
      o.x += g * yv.x; o.y += g * yv.y; o.z += g * yv.z; o.w += g * yv.w;
      *dst = o;
    }
  }
}

// final = LN(combined @ cmb_w^T + cmb_b) in place over the final region.
__global__ __launch_bounds__(TPB) void combine_kernel(
    float* __restrict__ out, const _Float16* __restrict__ cwh,
    const float* __restrict__ cb, const float* __restrict__ cg,
    const float* __restrict__ cbeta) {
  extern __shared__ char smem[];
  _Float16* sh_h = (_Float16*)smem;             // [64][256] f16
  _Float16* sh_b = sh_h + MTILE * HDIM;         // 2 x [128][256] f16
  float*    sh_y = (float*)(sh_b + 2 * BUFH);   // [64][256] f32
  float*    sh_m = sh_y + MTILE * HDIM;         // [64]
  float*    sh_v = sh_m + MTILE;                // [64]
  const int tid = threadIdx.x;
  const int lane = tid & 31;
  const int m0 = (tid >> 5) * 16;
  const long tok0 = (long)blockIdx.x * MTILE;
  const v8f vzero = {0.f, 0.f, 0.f, 0.f, 0.f, 0.f, 0.f, 0.f};

  stage_copy(sh_b, cwh, BUFH, tid);
  stage_copy(sh_b + BUFH, cwh + BUFH, BUFH, tid);

  for (int i = tid * 4; i < MTILE * HDIM; i += TPB * 4) {
    float4 v = *(const float4*)(out + tok0 * HDIM + i);
    sh_h[i + 0] = (_Float16)v.x;
    sh_h[i + 1] = (_Float16)v.y;
    sh_h[i + 2] = (_Float16)v.z;
    sh_h[i + 3] = (_Float16)v.w;
  }

#pragma unroll
  for (int nb = 0; nb < 2; ++nb) {
    if (nb == 0) async_wait_32(); else async_wait_0();
    __syncthreads();
    const _Float16* buf = sh_b + nb * BUFH;
    v8f acc[8];
#pragma unroll
    for (int t = 0; t < 8; ++t) acc[t] = vzero;
    wmma_block(sh_h, m0, HDIM, 0, buf, 256, 8, 8, acc, lane);
#pragma unroll
    for (int t = 0; t < 8; ++t)
#pragma unroll
      for (int ii = 0; ii < 8; ++ii) {
        int m = m0 + ii + ((lane & 16) ? 8 : 0);
        int n = nb * 128 + t * 16 + (lane & 15);
        sh_y[m * HDIM + n] = acc[t][ii] + cb[n];
      }
    __syncthreads();
  }

  if (tid < MTILE) {
    const float* yr = sh_y + tid * HDIM;
    float s = 0.f;
    for (int j = 0; j < HDIM; ++j) s += yr[j];
    float mean = s * (1.0f / HDIM);
    float vs = 0.f;
    for (int j = 0; j < HDIM; ++j) { float d = yr[j] - mean; vs += d * d; }
    sh_m[tid] = mean;
    sh_v[tid] = rsqrtf(vs * (1.0f / HDIM) + 1e-5f);
  }
  __syncthreads();

  for (int i = tid * 4; i < MTILE * HDIM; i += TPB * 4) {
    int r = i >> 8;
    int c = i & (HDIM - 1);
    float mean = sh_m[r], inv = sh_v[r];
    float4 yv = *(const float4*)(sh_y + i);
    float4 o;
    o.x = (yv.x - mean) * inv * cg[c + 0] + cbeta[c + 0];
    o.y = (yv.y - mean) * inv * cg[c + 1] + cbeta[c + 1];
    o.z = (yv.z - mean) * inv * cg[c + 2] + cbeta[c + 2];
    o.w = (yv.w - mean) * inv * cg[c + 3] + cbeta[c + 3];
    *(float4*)(out + tok0 * HDIM + i) = o;
  }
}

// ---------------- host launcher ----------------
extern "C" void kernel_launch(void* const* d_in, const int* in_sizes, int n_in,
                              void* d_out, int out_size, void* d_ws, size_t ws_size,
                              hipStream_t stream) {
  const float* x      = (const float*)d_in[0];
  const float* gate_w = (const float*)d_in[1];
  const float* gate_b = (const float*)d_in[2];
  const int EBASE = 3, EN = 11;
  const float* cmb_w    = (const float*)d_in[EBASE + 6 * EN + 0];
  const float* cmb_b    = (const float*)d_in[EBASE + 6 * EN + 1];
  const float* cmb_g    = (const float*)d_in[EBASE + 6 * EN + 2];
  const float* cmb_beta = (const float*)d_in[EBASE + 6 * EN + 3];

  const int T = in_sizes[0] / HDIM;  // 65536 tokens
  float* outp   = (float*)d_out;
  float* finalp = outp;                            // [T,256]
  float* confp  = outp + (size_t)T * HDIM;         // [T]
  float* topgp  = confp + T;                       // [T,2]
  int*   idxp   = (int*)(topgp + 2 * (size_t)T);   // [T,2]

  // carve f16 weight images out of the workspace
  _Float16* wsh = (_Float16*)d_ws;
  _Float16* w1h[6]; _Float16* w2h[6]; _Float16* cw1h[6];
  int d1s[6], d1pads[6];
  size_t off = 0;
  for (int e = 0; e < 6; ++e) {
    d1s[e] = in_sizes[EBASE + EN * e + 1] / HDIM;
    d1pads[e] = (d1s[e] + 31) & ~31;
    w1h[e] = wsh + off; off += (size_t)d1pads[e] * HDIM;
    w2h[e] = wsh + off; off += (size_t)d1pads[e] * HDIM;
  }
  for (int e = 0; e < 6; ++e) { cw1h[e] = wsh + off; off += BUFH / 4; }  // 128x256 padded
  _Float16* cmbh = wsh + off;

  // 0) one-time per-call weight conversion to padded f16 (staging order)
  for (int e = 0; e < 6; ++e) {
    const float* w1  = (const float*)d_in[EBASE + EN * e + 1];
    const float* w2  = (const float*)d_in[EBASE + EN * e + 3];
    const float* cw1 = (const float*)d_in[EBASE + EN * e + 7];
    prep_w1_kernel<<<128, 256, 0, stream>>>(w1, w1h[e], d1s[e], d1pads[e]);
    prep_w2_kernel<<<128, 256, 0, stream>>>(w2, w2h[e], d1s[e], d1pads[e]);
    prep_w1_kernel<<<32, 256, 0, stream>>>(cw1, cw1h[e], HDIM / 4, 128);
  }
  prep_w1_kernel<<<128, 256, 0, stream>>>(cmb_w, cmbh, HDIM, HDIM);

  // 1) zero accumulation regions (final + conf are contiguous)
  zero_kernel<<<2048, 256, 0, stream>>>(finalp, (size_t)T * HDIM + T);

  // 2) gating: one wave per token (8 waves per 256-thread block)
  gate_kernel<<<T / 8, 256, 0, stream>>>(x, gate_w, gate_b, topgp, idxp, T);

  // 3) six fused dense experts (stream-ordered accumulation into final/conf)
  for (int e = 0; e < 6; ++e) {
    const float* spec = (const float*)d_in[EBASE + EN * e + 0];
    const float* b1   = (const float*)d_in[EBASE + EN * e + 2];
    const float* b2   = (const float*)d_in[EBASE + EN * e + 4];
    const float* lng  = (const float*)d_in[EBASE + EN * e + 5];
    const float* lnb  = (const float*)d_in[EBASE + EN * e + 6];
    const float* cb1  = (const float*)d_in[EBASE + EN * e + 8];
    const float* cw2  = (const float*)d_in[EBASE + EN * e + 9];
    const float* cb2  = (const float*)d_in[EBASE + EN * e + 10];
    const size_t smem_expert =
        (size_t)(MTILE * HDIM + MTILE * d1pads[e] + 2 * BUFH) * sizeof(_Float16) +
        (size_t)(MTILE * HDIM + MTILE) * sizeof(float);
#define ARGS x, topgp, idxp, spec, w1h[e], b1, w2h[e], b2, lng, lnb, cw1h[e], \
             cb1, cw2, cb2, finalp, confp, d1s[e], e
    if (d1pads[e] == 512)
      expert_kernel<512, 0><<<T / MTILE, TPB, smem_expert, stream>>>(ARGS);
    else if (d1pads[e] == 416)
      expert_kernel<416, 1><<<T / MTILE, TPB, smem_expert, stream>>>(ARGS);
    else if (e == 2)
      expert_kernel<256, 2><<<T / MTILE, TPB, smem_expert, stream>>>(ARGS);
    else
      expert_kernel<256, 3><<<T / MTILE, TPB, smem_expert, stream>>>(ARGS);
#undef ARGS
  }

  // 4) final = LN(combined @ cmb_w^T + cmb_b), in place
  const size_t smem_combine =
      (size_t)(MTILE * HDIM + 2 * BUFH) * sizeof(_Float16) +
      (size_t)(MTILE * HDIM + 2 * MTILE) * sizeof(float);
  combine_kernel<<<T / MTILE, TPB, smem_combine, stream>>>(
      finalp, cmbh, cmb_b, cmb_g, cmb_beta);
}
